// GraphRNNDecoder_85349590106385
// MI455X (gfx1250) — matrix operand order repaired and checked by reading, not compiled, not run on hardware
//
#include <hip/hip_runtime.h>

// ---------------------------------------------------------------------------
// GraphRNN LSTM decoder rollout for gfx1250 (MI455X), wave32 + WMMA f16.
// N=4096 rows, T=128 sequential steps, H=256, D=4.
// Persistent WG owns 32 rows for the whole rollout; c-state in VGPRs;
// all matmuls via v_wmma_f32_16x16x32_f16 with augmented-K (K=288) packing
// that folds W_ih, biases, and the fc3 residual into the weight matrices.
// Elementwise LSTM uses native TRANS ops (v_tanh_f32 / v_exp_f32 / v_rcp_f32)
// to keep the latency-critical per-step path branch-free and short.
// ---------------------------------------------------------------------------

#define NN 4096
#define TT 128
#define DD 4
#define HH 256
#define GG 1024          // 4*H
#define KP 288           // padded K: 256 (h/p) + 4 (prev) + 1 (bias) + pad, = 9*32
#define KC 9             // K chunks of 32

typedef _Float16 v16h __attribute__((ext_vector_type(16)));
typedef _Float16 v8h  __attribute__((ext_vector_type(8)));
typedef float    v8f  __attribute__((ext_vector_type(8)));

// A-matrix 16x32 f16 fragment (ISA 7.12.2): lane = M row (lane&15);
// lane<16 holds K = {0..7, 16..23}, lane>=16 holds K = {8..15, 24..31}.
__device__ __forceinline__ v16h frag_a(const _Float16* rowp, int kk, int lh) {
    const _Float16* q = rowp + kk * 32 + lh * 8;
    v8h lo = *(const v8h*)(q);
    v8h hi = *(const v8h*)(q + 16);
    v16h r;
#pragma unroll
    for (int i = 0; i < 8; ++i) { r[i] = lo[i]; r[i + 8] = hi[i]; }
    return r;
}

// B-matrix 32x16 f16 fragment: lane = N column (lane&15);
// lane<16 holds K = {0..7, 16..23}, lane>=16 holds K = {8..15, 24..31} (matches A).
// For D = A * W^T, column n of B is row (colbase+n) of row-major W.
__device__ __forceinline__ v16h frag_b(const _Float16* rowp, int kk, int lh) {
    const _Float16* q = rowp + kk * 32 + lh * 8;
    v8h lo = *(const v8h*)(q);
    v8h hi = *(const v8h*)(q + 16);
    v16h r;
#pragma unroll
    for (int i = 0; i < 8; ++i) { r[i] = lo[i]; r[i + 8] = hi[i]; }
    return r;
}

__device__ __forceinline__ v8f wmma16(v16h a, v16h b, v8f c) {
    return __builtin_amdgcn_wmma_f32_16x16x32_f16(false, a, false, b,
                                                  (short)0, c, false, false);
}

// ---- branch-free transcendentals on the TRANS pipe ----
__device__ __forceinline__ float fast_tanh(float x) {
#if __has_builtin(__builtin_amdgcn_tanhf)
    return __builtin_amdgcn_tanhf(x);           // v_tanh_f32
#else
    // tanh(x) = 1 - 2/(exp(2x)+1), branch-free via v_exp_f32 + v_rcp_f32.
    float e = __expf(2.0f * x);
    return 1.0f - 2.0f * __builtin_amdgcn_rcpf(e + 1.0f);
#endif
}

__device__ __forceinline__ float fast_sig(float x) {
    // sigmoid(x) = 1/(1+exp(-x)): v_exp_f32 + v_rcp_f32, no divergence.
    return __builtin_amdgcn_rcpf(1.0f + __expf(-x));
}

// ---------------------------------------------------------------------------
// Prep: build packed f16 weight matrices in workspace (row-major, stride KP).
//   Wg [1024][KP]: [W_hh | W_ih | b_ih+b_hh | 0]
//   F1 [ 256][KP]: [fc1_w | 0 0 0 0 | fc1_b | 0]
//   F2 [ 256][KP]: [fc2_w | 0 0 0 0 | fc2_b | 0]
//   F3 [  16][KP]: rows 0..3 = [fc3_w | e_j (residual +prev) | fc3_b | 0]
// ---------------------------------------------------------------------------
__global__ void prep_weights(const float* __restrict__ W_ih, const float* __restrict__ W_hh,
                             const float* __restrict__ b_ih, const float* __restrict__ b_hh,
                             const float* __restrict__ fc1_w, const float* __restrict__ fc1_b,
                             const float* __restrict__ fc2_w, const float* __restrict__ fc2_b,
                             const float* __restrict__ fc3_w, const float* __restrict__ fc3_b,
                             _Float16* __restrict__ ws) {
    int r = blockIdx.x;
    int k = threadIdx.x;
    if (k >= KP) return;
    float v = 0.0f;
    if (r < 1024) {
        if (k < 256)       v = W_hh[r * 256 + k];
        else if (k < 260)  v = W_ih[r * 4 + (k - 256)];
        else if (k == 260) v = b_ih[r] + b_hh[r];
        ws[(size_t)r * KP + k] = (_Float16)v;
    } else if (r < 1280) {
        int j = r - 1024;
        if (k < 256)       v = fc1_w[j * 256 + k];
        else if (k == 260) v = fc1_b[j];
        ws[(size_t)(1024 + j) * KP + k] = (_Float16)v;
    } else if (r < 1536) {
        int j = r - 1280;
        if (k < 256)       v = fc2_w[j * 256 + k];
        else if (k == 260) v = fc2_b[j];
        ws[(size_t)(1280 + j) * KP + k] = (_Float16)v;
    } else {
        int j = r - 1536;
        if (j < DD) {
            if (k < 256)            v = fc3_w[j * 256 + k];
            else if (k == 256 + j)  v = 1.0f;    // + prev residual
            else if (k == 260)      v = fc3_b[j];
        }
        ws[(size_t)(1536 + j) * KP + k] = (_Float16)v;
    }
}

// ---------------------------------------------------------------------------
// Main rollout. grid = 128 WGs * 32 rows; block = 256 threads = 8 waves.
// Wave w owns columns [32w, 32w+32) of each gate quadrant / hidden vector.
// X buffer: [h | prev | 1 | 0]  (gates + fc1 input),  Y: [p | prev | 1 | 0].
// ---------------------------------------------------------------------------
__global__ __launch_bounds__(256) void lstm_rollout(
        const float* __restrict__ inputs,
        const _Float16* __restrict__ Wg, const _Float16* __restrict__ F1,
        const _Float16* __restrict__ F2, const _Float16* __restrict__ F3,
        float* __restrict__ out) {
    __shared__ __align__(16) _Float16 X[2][16][KP];
    __shared__ __align__(16) _Float16 Y[2][16][KP];

    const int tid = threadIdx.x;
    const int w   = tid >> 5;
    const int lane = tid & 31;
    const int lh  = lane >> 4;     // lane half
    const int ln  = lane & 15;     // M (for A) / N (for B/D)
    const int rowbase = blockIdx.x * 32;
    const int wid = __builtin_amdgcn_readfirstlane(w);

    // ---- init: h0 = 0, prev = inputs[:,0,:], bias column = 1, pads = 0 ----
    for (int idx = tid; idx < 2 * 16 * KP; idx += 256) {
        int rt  = idx / (16 * KP);
        int rem = idx - rt * 16 * KP;
        int row = rem / KP;
        int k   = rem - row * KP;
        _Float16 v = (_Float16)0.0f;
        if (k == 260) v = (_Float16)1.0f;
        else if (k >= 256 && k < 260)
            v = (_Float16)inputs[(size_t)(rowbase + rt * 16 + row) * TT * DD + (k - 256)];
        X[rt][row][k] = v;
        Y[rt][row][k] = v;
    }
    __syncthreads();

    v8f cst[2][2] = {};  // c-state: [rowtile][subtile], col = 32w+16s+ln, row = r+8*lh

#pragma unroll 1
    for (int t = 0; t < TT; ++t) {
        // ============ gates = [h|prev|1] @ [W_hh|W_ih|b]^T ============
        v8f acc[2][4][2] = {};
#pragma unroll 1
        for (int kk = 0; kk < KC; ++kk) {
            v16h a0 = frag_a(&X[0][ln][0], kk, lh);
            v16h a1 = frag_a(&X[1][ln][0], kk, lh);
#pragma unroll
            for (int q = 0; q < 4; ++q) {
#pragma unroll
                for (int s = 0; s < 2; ++s) {
                    v16h b = frag_b(Wg + (size_t)(q * 256 + w * 32 + s * 16 + ln) * KP, kk, lh);
                    acc[0][q][s] = wmma16(a0, b, acc[0][q][s]);
                    acc[1][q][s] = wmma16(a1, b, acc[1][q][s]);
                }
            }
        }
        __syncthreads();  // all X reads done before h_new overwrite

        // ======== LSTM cell update: register-local, TRANS-pipe only =========
#pragma unroll
        for (int rt = 0; rt < 2; ++rt) {
#pragma unroll
            for (int s = 0; s < 2; ++s) {
#pragma unroll
                for (int r = 0; r < 8; ++r) {
                    float ig = fast_sig(acc[rt][0][s][r]);
                    float fg = fast_sig(acc[rt][1][s][r]);
                    float gg = fast_tanh(acc[rt][2][s][r]);
                    float og = fast_sig(acc[rt][3][s][r]);
                    float cn = fg * cst[rt][s][r] + ig * gg;
                    cst[rt][s][r] = cn;
                    float hn = og * fast_tanh(cn);
                    X[rt][r + 8 * lh][w * 32 + s * 16 + ln] = (_Float16)hn;
                }
            }
        }
        __syncthreads();

        // ============ fc1: Y = relu([h_new|.|1] @ F1^T) ============
        {
            v8f p[2][2] = {};
#pragma unroll 1
            for (int kk = 0; kk < KC; ++kk) {
                v16h a0 = frag_a(&X[0][ln][0], kk, lh);
                v16h a1 = frag_a(&X[1][ln][0], kk, lh);
#pragma unroll
                for (int s = 0; s < 2; ++s) {
                    v16h b = frag_b(F1 + (size_t)(w * 32 + s * 16 + ln) * KP, kk, lh);
                    p[0][s] = wmma16(a0, b, p[0][s]);
                    p[1][s] = wmma16(a1, b, p[1][s]);
                }
            }
            // no prior Y readers outstanding (barrier after fc3 last step)
#pragma unroll
            for (int rt = 0; rt < 2; ++rt)
#pragma unroll
                for (int s = 0; s < 2; ++s)
#pragma unroll
                    for (int r = 0; r < 8; ++r)
                        Y[rt][r + 8 * lh][w * 32 + s * 16 + ln] =
                            (_Float16)fmaxf(p[rt][s][r], 0.0f);
        }
        __syncthreads();

        // ============ fc2: Y = relu([p1|.|1] @ F2^T)  (in-place via barrier) ==
        {
            v8f p[2][2] = {};
#pragma unroll 1
            for (int kk = 0; kk < KC; ++kk) {
                v16h a0 = frag_a(&Y[0][ln][0], kk, lh);
                v16h a1 = frag_a(&Y[1][ln][0], kk, lh);
#pragma unroll
                for (int s = 0; s < 2; ++s) {
                    v16h b = frag_b(F2 + (size_t)(w * 32 + s * 16 + ln) * KP, kk, lh);
                    p[0][s] = wmma16(a0, b, p[0][s]);
                    p[1][s] = wmma16(a1, b, p[1][s]);
                }
            }
            __syncthreads();  // all Y reads complete before overwrite
#pragma unroll
            for (int rt = 0; rt < 2; ++rt)
#pragma unroll
                for (int s = 0; s < 2; ++s)
#pragma unroll
                    for (int r = 0; r < 8; ++r)
                        Y[rt][r + 8 * lh][w * 32 + s * 16 + ln] =
                            (_Float16)fmaxf(p[rt][s][r], 0.0f);
        }
        __syncthreads();

        // ===== fc3: p = [p2|prev|1] @ F3^T (residual+bias folded into F3) =====
        if (wid < 2) {  // wave `wid` handles row-tile `wid`
            v8f p3 = {};
#pragma unroll 1
            for (int kk = 0; kk < KC; ++kk) {
                v16h a = frag_a(&Y[wid][ln][0], kk, lh);
                v16h b = frag_b(F3 + (size_t)ln * KP, kk, lh);
                p3 = wmma16(a, b, p3);
            }
            if (ln < DD) {
#pragma unroll
                for (int r = 0; r < 8; ++r) {
                    int row = r + 8 * lh;
                    float v = p3[r];
                    out[(size_t)(rowbase + wid * 16 + row) * TT * DD + (size_t)t * DD + ln] = v;
                    _Float16 hv = (_Float16)v;   // prev_{t+1} for gates and fc3 residual
                    X[wid][row][256 + ln] = hv;
                    Y[wid][row][256 + ln] = hv;
                }
            }
        }
        __syncthreads();
    }
}

// ---------------------------------------------------------------------------
extern "C" void kernel_launch(void* const* d_in, const int* in_sizes, int n_in,
                              void* d_out, int out_size, void* d_ws, size_t ws_size,
                              hipStream_t stream) {
    const float* inputs = (const float*)d_in[0];
    const float* W_ih   = (const float*)d_in[1];
    const float* W_hh   = (const float*)d_in[2];
    const float* b_ih   = (const float*)d_in[3];
    const float* b_hh   = (const float*)d_in[4];
    const float* fc1_w  = (const float*)d_in[5];
    const float* fc1_b  = (const float*)d_in[6];
    const float* fc2_w  = (const float*)d_in[7];
    const float* fc2_b  = (const float*)d_in[8];
    const float* fc3_w  = (const float*)d_in[9];
    const float* fc3_b  = (const float*)d_in[10];

    _Float16* ws = (_Float16*)d_ws;  // 1552*288*2 = ~894 KB of packed f16 weights
    _Float16* Wg = ws;
    _Float16* F1 = ws + (size_t)1024 * KP;
    _Float16* F2 = ws + (size_t)1280 * KP;
    _Float16* F3 = ws + (size_t)1536 * KP;

    prep_weights<<<1552, KP, 0, stream>>>(W_ih, W_hh, b_ih, b_hh, fc1_w, fc1_b,
                                          fc2_w, fc2_b, fc3_w, fc3_b, ws);
    lstm_rollout<<<NN / 32, 256, 0, stream>>>(inputs, Wg, F1, F2, F3, (float*)d_out);
}
